// Attention_5042291605946
// MI455X (gfx1250) — matrix operand order repaired and checked by reading, not compiled, not run on hardware
//
#include <hip/hip_runtime.h>

#define BB 2
#define SS 2048
#define DD 4096
#define HH 32
#define HDD 128

typedef __attribute__((ext_vector_type(16))) __bf16          v16bf;
typedef __attribute__((ext_vector_type(16))) unsigned short  v16u;
typedef __attribute__((ext_vector_type(8)))  unsigned short  v8u;
typedef __attribute__((ext_vector_type(8)))  float           v8f;
typedef __attribute__((ext_vector_type(4)))  int             v4i;

static __device__ __forceinline__ unsigned short f2bf(float f) {
  unsigned int u = __float_as_uint(f);
  u += 0x7fffu + ((u >> 16) & 1u);            // round-to-nearest-even
  return (unsigned short)(u >> 16);
}
static __device__ __forceinline__ float bf2f(unsigned short h) {
  return __uint_as_float(((unsigned int)h) << 16);
}

static __device__ __forceinline__ v8f wmma_bf16(v16u a, v16u b, v8f c) {
  return __builtin_amdgcn_wmma_f32_16x16x32_bf16(
      false, __builtin_bit_cast(v16bf, a),
      false, __builtin_bit_cast(v16bf, b),
      (short)0, c, false, false);
}

// A fragment (16x32 bf16): rows m0..m0+15, K-slice [k0,k0+32).
// lane<16: K 0..7 & 16..23 of row (lane&15); lane>=16: K 8..15 & 24..31.
static __device__ __forceinline__ v16u load_a(const unsigned short* base, long ldm,
                                              int m0, int k0, int lane) {
  const unsigned short* p =
      base + (long)(m0 + (lane & 15)) * ldm + k0 + ((lane >> 4) & 1) * 8;
  v8u lo = *(const v8u*)p;
  v8u hi = *(const v8u*)(p + 16);
  v16u r;
#pragma unroll
  for (int i = 0; i < 8; ++i) { r[i] = lo[i]; r[i + 8] = hi[i]; }
  return r;
}

// B fragment (32x16 bf16): element B[k][n] = base[n*ldn + k].
// lane<16: K 0..7 & 16..23 of column (lane&15); lane>=16: K 8..15 & 24..31.
static __device__ __forceinline__ v16u load_b(const unsigned short* base, long ldn,
                                              int n0, int k0, int lane) {
  const unsigned short* p =
      base + (long)(n0 + (lane & 15)) * ldn + k0 + ((lane >> 4) & 1) * 8;
  v8u lo = *(const v8u*)p;
  v8u hi = *(const v8u*)(p + 16);
  v16u r;
#pragma unroll
  for (int i = 0; i < 8; ++i) { r[i] = lo[i]; r[i + 8] = hi[i]; }
  return r;
}

// ------------------------------------------------ CDNA5 async copy primitives
static __device__ __forceinline__ void async_copy16(const unsigned short* g,
                                                    unsigned short* l) {
#if __has_builtin(__builtin_amdgcn_global_load_async_to_lds_b128)
  __builtin_amdgcn_global_load_async_to_lds_b128(
      (__attribute__((address_space(1))) v4i*)(unsigned short*)g,
      (__attribute__((address_space(3))) v4i*)l,
      0, 0);
#else
  *(v8u*)l = *(const v8u*)g;   // synchronous fallback
#endif
}

static __device__ __forceinline__ void wait_async_keep3(void) {
#if __has_builtin(__builtin_amdgcn_s_wait_asynccnt)
  __builtin_amdgcn_s_wait_asynccnt(3);
#else
  asm volatile("s_wait_asynccnt 0x3" ::: "memory");
#endif
}
static __device__ __forceinline__ void wait_async_zero(void) {
#if __has_builtin(__builtin_amdgcn_s_wait_asynccnt)
  __builtin_amdgcn_s_wait_asynccnt(0);
#else
  asm volatile("s_wait_asynccnt 0x0" ::: "memory");
#endif
}

// ---------------------------------------------------------------- fp32 -> bf16
__global__ __launch_bounds__(256) void cvt_bf16(const float* __restrict__ src,
                                                unsigned short* __restrict__ dst,
                                                long n) {
  long i = (long)blockIdx.x * blockDim.x + threadIdx.x;
  if (i < n) dst[i] = f2bf(src[i]);
}

// ---------------------------------------------------------------- bf16 GEMM
// C[m,n] = sum_k A[m,k] * W[n,k]   (A: MxK row-major, W: NxK row-major)
// Block = 8 waves = 128x64 tile, K-step 32. A/B tiles double-buffered in LDS,
// filled with GLOBAL_LOAD_ASYNC_TO_LDS_B128 (ASYNCcnt) each K-step.
// Rows padded to 40 ushorts (80 B) for bank-conflict-free ds_load_b128.
// mode 0: bf16 row-major out  (Q,K projections)
// mode 1: bf16 per-head transposed V out: vt[(b*H+h)*HD + d][s]
// mode 2: fp32 row-major out  (final projection)
#define LDA 40
__global__ __launch_bounds__(256) void gemm_bf16(const unsigned short* __restrict__ A,
                                                 const unsigned short* __restrict__ W,
                                                 void* __restrict__ C,
                                                 int N, int K, int mode) {
  __shared__ __align__(16) unsigned short sA[2][128 * LDA];
  __shared__ __align__(16) unsigned short sB[2][64 * LDA];

  const int tid  = threadIdx.x;
  const int lane = tid & 31;
  const int wave = tid >> 5;
  const int m0 = blockIdx.x * 128 + wave * 16;
  const int n0 = blockIdx.y * 64;

  v8f acc[4];
#pragma unroll
  for (int t = 0; t < 4; ++t)
#pragma unroll
    for (int j = 0; j < 8; ++j) acc[t][j] = 0.0f;

  const int mblk = blockIdx.x * 128;
  // stage K-slice [k0,k0+32) of the block's A (128 rows) and B (64 rows) tiles
  auto stage = [&](int k0, int buf) {
#pragma unroll
    for (int j = 0; j < 2; ++j) {           // A: 512 b128 chunks, 2 per thread
      int c = tid + 256 * j;
      int row = c >> 2, seg = c & 3;
      async_copy16(A + (long)(mblk + row) * K + k0 + seg * 8,
                   &sA[buf][row * LDA + seg * 8]);
    }
    {                                        // B: 256 b128 chunks, 1 per thread
      int row = tid >> 2, seg = tid & 3;
      async_copy16(W + (long)(n0 + row) * K + k0 + seg * 8,
                   &sB[buf][row * LDA + seg * 8]);
    }
  };

  stage(0, 0);
  for (int k0 = 0; k0 < K; k0 += 32) {
    const int cur = (k0 >> 5) & 1;
    const bool more = (k0 + 32) < K;
    if (more) stage(k0 + 32, cur ^ 1);       // prefetch next K-slice
    if (more) wait_async_keep3();            // our 3 current-slice copies done
    else      wait_async_zero();
    __syncthreads();                         // everyone's copies landed

    v16u af = load_a(&sA[cur][0], LDA, wave * 16, 0, lane);
#pragma unroll
    for (int t = 0; t < 4; ++t) {
      v16u bf = load_b(&sB[cur][0], LDA, t * 16, 0, lane);
      acc[t] = wmma_bf16(af, bf, acc[t]);
    }
    __syncthreads();                         // all reads done before overwrite
  }

  const int sel = (lane >> 4) & 1;
  const int lc  = lane & 15;
  if (mode == 2) {
    float* Cf = (float*)C;
#pragma unroll
    for (int t = 0; t < 4; ++t)
#pragma unroll
      for (int j = 0; j < 8; ++j) {
        int row = m0 + j + sel * 8, col = n0 + t * 16 + lc;
        Cf[(long)row * N + col] = acc[t][j];
      }
  } else if (mode == 0) {
    unsigned short* Cb = (unsigned short*)C;
#pragma unroll
    for (int t = 0; t < 4; ++t)
#pragma unroll
      for (int j = 0; j < 8; ++j) {
        int row = m0 + j + sel * 8, col = n0 + t * 16 + lc;
        Cb[(long)row * N + col] = f2bf(acc[t][j]);
      }
  } else {  // mode 1: V transposed per head
    unsigned short* Cb = (unsigned short*)C;
#pragma unroll
    for (int t = 0; t < 4; ++t)
#pragma unroll
      for (int j = 0; j < 8; ++j) {
        int row = m0 + j + sel * 8;          // token index in [0, B*S)
        int col = n0 + t * 16 + lc;          // feature index = h*HD + d
        int b = row / SS, s = row % SS;
        int h = col >> 7, d = col & (HDD - 1);
        Cb[(((long)(b * HH + h) * HDD + d) * SS) + s] = f2bf(acc[t][j]);
      }
  }
}

// ---------------------------------------------------------------- RoPE (Q,K in place)
__global__ __launch_bounds__(256) void rope_qk(unsigned short* __restrict__ q,
                                               unsigned short* __restrict__ k,
                                               const float* __restrict__ fc,
                                               const float* __restrict__ fs) {
  long idx = (long)blockIdx.x * blockDim.x + threadIdx.x;  // over B*S*H*(HD/2)
  if (idx >= (long)BB * SS * HH * (HDD / 2)) return;
  int i = idx & 63;
  long r = idx >> 6;
  int h = r & (HH - 1);
  long t = r >> 5;                 // b*S + s
  int s = (int)(t % SS);
  float c  = fc[s * (HDD / 2) + i];
  float sn = fs[s * (HDD / 2) + i];
  long pos = t * DD + h * HDD + 2 * i;
  {
    float x0 = bf2f(q[pos]), x1 = bf2f(q[pos + 1]);
    q[pos]     = f2bf(x0 * c - x1 * sn);
    q[pos + 1] = f2bf(x0 * sn + x1 * c);
  }
  {
    float x0 = bf2f(k[pos]), x1 = bf2f(k[pos + 1]);
    k[pos]     = f2bf(x0 * c - x1 * sn);
    k[pos + 1] = f2bf(x0 * sn + x1 * c);
  }
}

// ---------------------------------------------------------------- flash attention
// One wave per 16-query tile per (b,h). Key tiles of 32, online softmax.
__global__ __launch_bounds__(256) void flash_attn(const unsigned short* __restrict__ qb,
                                                  const unsigned short* __restrict__ kb,
                                                  const unsigned short* __restrict__ vt,
                                                  unsigned short* __restrict__ ab) {
  __shared__ __align__(16) unsigned short pb[8][16 * 32];  // per-wave P tile

  const int lane = threadIdx.x & 31;
  const int wave = threadIdx.x >> 5;
  const int b = blockIdx.z, h = blockIdx.y;
  const int q0 = (blockIdx.x * 8 + wave) * 16;
  const int sel = (lane >> 4) & 1;
  const int lc  = lane & 15;

  const unsigned short* qbase = qb + (long)b * SS * DD + h * HDD;   // row stride DD
  const unsigned short* kbase = kb + (long)b * SS * DD + h * HDD;   // row stride DD
  const unsigned short* vbase = vt + (long)(b * HH + h) * HDD * SS; // row stride SS

  // Q A-fragments for the 4 head-dim chunks (K = 0..127 step 32)
  v16u qa[4];
#pragma unroll
  for (int c = 0; c < 4; ++c) qa[c] = load_a(qbase, DD, q0, c * 32, lane);

  v8f o[8];
  float m[8], l[8];
#pragma unroll
  for (int j = 0; j < 8; ++j) {
    m[j] = -1e30f; l[j] = 0.0f;
#pragma unroll
    for (int t = 0; t < 8; ++t) o[t][j] = 0.0f;
  }

  const float scale = 0.08838834764831845f;  // 1/sqrt(128)
  const int kmax = q0 + 16;                  // causal bound

  for (int k0 = 0; k0 < kmax; k0 += 32) {
    // scores: two 16x16 tiles (keys k0..k0+15 and k0+16..k0+31)
    v8f s0, s1;
#pragma unroll
    for (int j = 0; j < 8; ++j) { s0[j] = 0.0f; s1[j] = 0.0f; }
#pragma unroll
    for (int c = 0; c < 4; ++c) {
      v16u b0 = load_b(kbase, DD, k0, c * 32, lane);
      v16u b1 = load_b(kbase, DD, k0 + 16, c * 32, lane);
      s0 = wmma_bf16(qa[c], b0, s0);
      s1 = wmma_bf16(qa[c], b1, s1);
    }

    float alpha[8];
#pragma unroll
    for (int j = 0; j < 8; ++j) {
      int qr = q0 + j + sel * 8;
      int kc = k0 + lc;
      s0[j] = s0[j] * scale + ((kc      > qr) ? -1e9f : 0.0f);
      s1[j] = s1[j] * scale + ((kc + 16 > qr) ? -1e9f : 0.0f);
      // row max over 32 keys (16-lane halves reduce independently)
      float mx = fmaxf(s0[j], s1[j]);
#pragma unroll
      for (int d = 1; d < 16; d <<= 1) mx = fmaxf(mx, __shfl_xor(mx, d, 32));
      float nm = fmaxf(m[j], mx);
      float al = __expf(m[j] - nm);
      m[j] = nm;
      s0[j] = __expf(s0[j] - nm);
      s1[j] = __expf(s1[j] - nm);
      float sm = s0[j] + s1[j];
#pragma unroll
      for (int d = 1; d < 16; d <<= 1) sm += __shfl_xor(sm, d, 32);
      l[j] = l[j] * al + sm;
      alpha[j] = al;
    }
#pragma unroll
    for (int t = 0; t < 8; ++t)
#pragma unroll
      for (int j = 0; j < 8; ++j) o[t][j] *= alpha[j];

    // P (D-layout) -> LDS -> A-layout fragment
#pragma unroll
    for (int j = 0; j < 8; ++j) {
      int r = j + sel * 8;
      pb[wave][r * 32 + lc]      = f2bf(s0[j]);
      pb[wave][r * 32 + lc + 16] = f2bf(s1[j]);
    }
    v16u pa = load_a(&pb[wave][0], 32, 0, 0, lane);

    // O += P * V   (8 column tiles of 16 over head dim)
#pragma unroll
    for (int t = 0; t < 8; ++t) {
      v16u bv = load_b(vbase, SS, t * 16, k0, lane);  // B[k=key][n=d] = vt[d][key]
      o[t] = wmma_bf16(pa, bv, o[t]);
    }
  }

  // normalize and store bf16 (token-major for the final GEMM)
  unsigned short* abase = ab + (long)b * SS * DD + h * HDD;
#pragma unroll
  for (int j = 0; j < 8; ++j) {
    float inv = 1.0f / l[j];
    int row = q0 + j + sel * 8;
#pragma unroll
    for (int t = 0; t < 8; ++t)
      abase[(long)row * DD + t * 16 + lc] = f2bf(o[t][j] * inv);
  }
}

// ---------------------------------------------------------------- launcher
extern "C" void kernel_launch(void* const* d_in, const int* in_sizes, int n_in,
                              void* d_out, int out_size, void* d_ws, size_t ws_size,
                              hipStream_t stream) {
  (void)in_sizes; (void)n_in; (void)out_size; (void)ws_size;
  const float* x  = (const float*)d_in[0];
  const float* fc = (const float*)d_in[1];
  const float* fs = (const float*)d_in[2];
  // d_in[3] (mask) unused: causal mask computed analytically (start_pos == 0)
  const float* wq = (const float*)d_in[4];
  const float* wk = (const float*)d_in[5];
  const float* wv = (const float*)d_in[6];
  const float* wo = (const float*)d_in[7];

  const long E = (long)BB * SS * DD;  // == DD*DD == 16,777,216 elements
  unsigned short* p   = (unsigned short*)d_ws;
  unsigned short* xb  = p;          // x bf16
  unsigned short* wqb = p + 1 * E;
  unsigned short* wkb = p + 2 * E;
  unsigned short* wvb = p + 3 * E;
  unsigned short* wob = p + 4 * E;
  unsigned short* qbb = p + 5 * E;  // Q (bf16, row-major)
  unsigned short* kbb = p + 6 * E;  // K (bf16, row-major)
  unsigned short* vtb = p + 7 * E;  // V (bf16, per-head transposed [b][h][d][s])
  unsigned short* abb = p + 8 * E;  // attention output (bf16, row-major)

  const int cvtBlocks = (int)((E + 255) / 256);
  cvt_bf16<<<cvtBlocks, 256, 0, stream>>>(x,  xb,  E);
  cvt_bf16<<<cvtBlocks, 256, 0, stream>>>(wq, wqb, E);
  cvt_bf16<<<cvtBlocks, 256, 0, stream>>>(wk, wkb, E);
  cvt_bf16<<<cvtBlocks, 256, 0, stream>>>(wv, wvb, E);
  cvt_bf16<<<cvtBlocks, 256, 0, stream>>>(wo, wob, E);

  const int M = BB * SS;  // 4096
  dim3 gg(M / 128, DD / 64);
  gemm_bf16<<<gg, 256, 0, stream>>>(xb, wqb, qbb, DD, DD, 0);
  gemm_bf16<<<gg, 256, 0, stream>>>(xb, wkb, kbb, DD, DD, 0);
  gemm_bf16<<<gg, 256, 0, stream>>>(xb, wvb, vtb, DD, DD, 1);

  const long nrope = (long)BB * SS * HH * (HDD / 2);
  rope_qk<<<(int)((nrope + 255) / 256), 256, 0, stream>>>(qbb, kbb, fc, fs);

  dim3 ga(SS / 16 / 8, HH, BB);
  flash_attn<<<ga, 256, 0, stream>>>(qbb, kbb, vtb, abb);

  gemm_bf16<<<gg, 256, 0, stream>>>(abb, wob, d_out, DD, DD, 2);
}
